// DNCWrapper_82635170775171
// MI455X (gfx1250) — compile-verified, matrix operand order checked by reading, not run on hardware
//
#include <hip/hip_runtime.h>
#include <hip/hip_bf16.h>
#include <math.h>

typedef __attribute__((ext_vector_type(16))) _Float16 v16h;
typedef __attribute__((ext_vector_type(8)))  float    v8f;

#define N_CELLS 10
#define CELL    20
#define HID     128
#define IN_DIM  10
#define OUT_DIM 10
#define IFACE   88
#define T_STEPS 512
#define BTILE   16

// packed-weight offsets in halfs inside d_ws
#define OFF_IH0 0          // 32 frags  (N=512 pad, K=30->32)
#define OFF_HH0 16384      // 128 frags (N=512, K=128)
#define OFF_IH1 81920      // 128 frags
#define OFF_HH1 147456     // 128 frags
#define OFF_IF  212992     // 24 frags  (N=88->96, K=128)

__device__ __forceinline__ float sigf(float x) { return 1.f / (1.f + __expf(-x)); }
__device__ __forceinline__ float spf(float x)  { return x > 20.f ? x : log1pf(__expf(x)); }

// ---------------- weight pre-pack: f32 (N,K) row-major -> WMMA B fragments (f16) ----
// fragment layout: dst[frag*512 + lane*16 + i], fragment (nt,kt) -> frag = nt*KT + kt
// lane holds column n = nt*16 + (lane&15); element i -> K per ISA 16-bit interleave.
__global__ void prepack_kernel(const float* __restrict__ W, _Float16* __restrict__ dst,
                               int N, int K, int NT, int KT) {
  int idx = blockIdx.x * blockDim.x + threadIdx.x;
  int total = NT * KT * 512;
  if (idx >= total) return;
  int frag = idx >> 9;
  int r    = idx & 511;
  int lane = r >> 4;
  int i    = r & 15;
  int nt = frag / KT, kt = frag % KT;
  int n  = nt * 16 + (lane & 15);
  int p  = i >> 1, lo = i & 1;
  int k  = kt * 32 + ((p < 4) ? 0 : 16) + ((lane < 16) ? 0 : 8) + 2 * (p & 3) + lo;
  float v = (n < N && k < K) ? W[(size_t)n * K + k] : 0.f;
  dst[idx] = (_Float16)v;
}

// ---------------- fragment loaders -------------------------------------------------
__device__ __forceinline__ v16h load_a(const _Float16* a16, int stride, int lane, int kbase) {
  int m  = lane & 15;
  int hb = (lane & 16) ? 8 : 0;
  const _Float16* row = a16 + (size_t)m * stride + kbase + hb;
  v16h a = {};
#pragma unroll
  for (int p = 0; p < 8; ++p) {
    int k = ((p < 4) ? 0 : 16) + 2 * (p & 3);
    a[2 * p]     = row[k];
    a[2 * p + 1] = row[k + 1];
  }
  return a;
}

__device__ __forceinline__ v16h load_b(const _Float16* base, int frag, int lane) {
  return *(const v16h*)(base + (((size_t)frag * 32 + lane) << 4));
}

__device__ __forceinline__ v8f wmma_f16(v16h a, v16h b, v8f c) {
  return __builtin_amdgcn_wmma_f32_16x16x32_f16(false, a, false, b, (short)0, c, false, false);
}

// ---------------- persistent DNC kernel --------------------------------------------
struct Smem {
  float    c0[BTILE][HID];
  float    c1[BTILE][HID];
  _Float16 h0h[BTILE][HID];     // h0 (f16 staging, also the carried state)
  _Float16 h1h[BTILE][HID];     // h1 (pre-clip)
  _Float16 ch [BTILE][HID];     // ctrl = clip(h1)
  _Float16 inph[BTILE][32];     // [x_t | rv | pad]
  float    xi [BTILE][96];      // interface vector (88 valid)
  float    mem[BTILE][N_CELLS][CELL];
  float    link[BTILE][N_CELLS][N_CELLS];
  float    usage[BTILE][N_CELLS];
  float    prec [BTILE][N_CELLS];
  float    rw   [BTILE][N_CELLS];
  float    ww   [BTILE][N_CELLS];
  float    rv[BTILE][CELL];
  float    er[BTILE][CELL];
  float    wvv[BTILE][CELL];
};

__global__ __launch_bounds__(256)
void dnc_persistent(const float* __restrict__ x,
                    const float* __restrict__ b0,
                    const float* __restrict__ b1,
                    const float* __restrict__ b_if,
                    const float* __restrict__ W_out,
                    const float* __restrict__ b_out,
                    const _Float16* __restrict__ wpack,
                    float* __restrict__ out) {
  __shared__ Smem S;
  const int tid   = threadIdx.x;
  const int wid   = tid >> 5;
  const int lane  = tid & 31;
  const int bbase = blockIdx.x * BTILE;

  {
    int* p = (int*)&S;
    int n = (int)(sizeof(Smem) / 4);
    for (int i = tid; i < n; i += 256) p[i] = 0;
  }
  __syncthreads();

  const int nl = lane & 15;
  const int mo = (lane & 16) ? 8 : 0;   // D layout: vgpr r -> M = r + mo
  const int j  = wid * 16 + nl;         // gate hidden index slice per wave
  const v8f vzero = {};

  // persisted across stage A -> stage B within the same (tid<16) thread
  float rk[CELL];
#pragma unroll
  for (int e = 0; e < CELL; ++e) rk[e] = 0.f;
  float rs = 0.f, md0 = 0.f, md1 = 0.f, md2 = 0.f;

  for (int t = 0; t < T_STEPS; ++t) {
    // Opaque *scalar offset* (not pointer) redefined every timestep:
    //  - defeats LICM so 56 B-fragments are re-streamed from L2 each step
    //    (no 448-VGPR hoist, no scratch spills), AND
    //  - keeps the address chain rooted in the kernarg pointer so the backend
    //    infers the global address space -> global_load_b128 (LOADcnt only),
    //    instead of flat_load_b128 which would also pollute DScnt and couple
    //    the weight stream with the LDS A-fragment waits.
    size_t woff = 0;
    asm volatile("" : "+s"(woff));
    const _Float16* wp_t = wpack + woff;
    const _Float16* Wih0 = wp_t + OFF_IH0;
    const _Float16* Whh0 = wp_t + OFF_HH0;
    const _Float16* Wih1 = wp_t + OFF_IH1;
    const _Float16* Whh1 = wp_t + OFF_HH1;
    const _Float16* Wifp = wp_t + OFF_IF;

    // ---- build controller input [x_t | rv] (f16) ----
    for (int idx = tid; idx < BTILE * 32; idx += 256) {
      int m = idx >> 5, c = idx & 31;
      float v = 0.f;
      if (c < IN_DIM)             v = x[((size_t)(bbase + m) * T_STEPS + t) * IN_DIM + c];
      else if (c < IN_DIM + CELL) v = S.rv[m][c - IN_DIM];
      S.inph[m][c] = (_Float16)v;
    }
    __syncthreads();

    // ======== LSTM layer 0: gates = inp@Wih0^T + h0@Whh0^T ========
    v8f acc[4];
#pragma unroll
    for (int g = 0; g < 4; ++g) acc[g] = vzero;
    {
      v16h ai = load_a(&S.inph[0][0], 32, lane, 0);
#pragma unroll
      for (int g = 0; g < 4; ++g)
        acc[g] = wmma_f16(ai, load_b(Wih0, g * 8 + wid, lane), acc[g]);
#pragma unroll
      for (int kt = 0; kt < 4; ++kt) {
        v16h ah = load_a(&S.h0h[0][0], HID, lane, kt * 32);
#pragma unroll
        for (int g = 0; g < 4; ++g)
          acc[g] = wmma_f16(ah, load_b(Whh0, (g * 8 + wid) * 4 + kt, lane), acc[g]);
      }
    }
    __syncthreads();
#pragma unroll
    for (int r = 0; r < 8; ++r) {
      int m = r + mo;
      float iv = acc[0][r] + b0[j];
      float fv = acc[1][r] + b0[128 + j];
      float gv = acc[2][r] + b0[256 + j];
      float ov = acc[3][r] + b0[384 + j];
      float c  = sigf(fv) * S.c0[m][j] + sigf(iv) * tanhf(gv);
      float h  = sigf(ov) * tanhf(c);
      S.c0[m][j]  = c;
      S.h0h[m][j] = (_Float16)h;
    }
    __syncthreads();

    // ======== LSTM layer 1: gates = h0@Wih1^T + h1@Whh1^T ========
#pragma unroll
    for (int g = 0; g < 4; ++g) acc[g] = vzero;
#pragma unroll
    for (int kt = 0; kt < 4; ++kt) {
      v16h ai = load_a(&S.h0h[0][0], HID, lane, kt * 32);
      v16h ah = load_a(&S.h1h[0][0], HID, lane, kt * 32);
#pragma unroll
      for (int g = 0; g < 4; ++g) {
        acc[g] = wmma_f16(ai, load_b(Wih1, (g * 8 + wid) * 4 + kt, lane), acc[g]);
        acc[g] = wmma_f16(ah, load_b(Whh1, (g * 8 + wid) * 4 + kt, lane), acc[g]);
      }
    }
    __syncthreads();
#pragma unroll
    for (int r = 0; r < 8; ++r) {
      int m = r + mo;
      float iv = acc[0][r] + b1[j];
      float fv = acc[1][r] + b1[128 + j];
      float gv = acc[2][r] + b1[256 + j];
      float ov = acc[3][r] + b1[384 + j];
      float c  = sigf(fv) * S.c1[m][j] + sigf(iv) * tanhf(gv);
      float h  = sigf(ov) * tanhf(c);
      S.c1[m][j]  = c;
      S.h1h[m][j] = (_Float16)h;
      S.ch[m][j]  = (_Float16)fminf(fmaxf(h, -20.f), 20.f);
    }
    __syncthreads();

    // ======== interface vector: xi = ctrl@Wif^T + b_if ========
    if (wid < 6) {
      v8f ax = vzero;
#pragma unroll
      for (int kt = 0; kt < 4; ++kt) {
        v16h a = load_a(&S.ch[0][0], HID, lane, kt * 32);
        ax = wmma_f16(a, load_b(Wifp, wid * 4 + kt, lane), ax);
      }
      int n   = wid * 16 + nl;
      float bb = (n < IFACE) ? b_if[n] : 0.f;
#pragma unroll
      for (int r = 0; r < 8; ++r) S.xi[r + mo][n] = ax[r] + bb;
    }
    __syncthreads();

    // ======== DNC memory machine, stage A (per batch-row thread) ========
    if (tid < BTILE) {
      int m = tid;
      const float* q = &S.xi[m][0];
      float wk[CELL];
#pragma unroll
      for (int e = 0; e < CELL; ++e) {
        rk[e]       = tanhf(q[e]);
        wk[e]       = tanhf(q[21 + e]);
        S.er[m][e]  = sigf(q[42 + e]);
        S.wvv[m][e] = tanhf(q[62 + e]);
      }
      rs = spf(q[20]);
      float wstr = spf(q[41]);
      float fg = sigf(q[82]), ga = sigf(q[83]), gwt = sigf(q[84]);
      { float a0 = q[85], a1 = q[86], a2 = q[87];
        float mx = fmaxf(a0, fmaxf(a1, a2));
        float e0 = __expf(a0 - mx), e1 = __expf(a1 - mx), e2 = __expf(a2 - mx);
        float es = e0 + e1 + e2; md0 = e0 / es; md1 = e1 / es; md2 = e2 / es; }
      // usage update: (u + ww - u*ww) * psi,  psi = 1 - fg*rw  (R=1)
      for (int n = 0; n < N_CELLS; ++n) {
        float psi = 1.f - fg * S.rw[m][n];
        float u = S.usage[m][n], w0 = S.ww[m][n];
        S.usage[m][n] = (u + w0 - u * w0) * psi;
      }
      // allocation weights via sort
      float su[N_CELLS]; int idn[N_CELLS];
      for (int n = 0; n < N_CELLS; ++n) { su[n] = 1e-6f + (1.f - 1e-6f) * S.usage[m][n]; idn[n] = n; }
      for (int i2 = 0; i2 < N_CELLS - 1; ++i2) {
        int mn = i2;
        for (int j2 = i2 + 1; j2 < N_CELLS; ++j2) if (su[j2] < su[mn]) mn = j2;
        float tv = su[i2]; su[i2] = su[mn]; su[mn] = tv;
        int   ti = idn[i2]; idn[i2] = idn[mn]; idn[mn] = ti;
      }
      float alloc[N_CELLS];
      { float excl = 1.f;
        for (int n = 0; n < N_CELLS; ++n) { alloc[idn[n]] = (1.f - su[n]) * excl; excl *= su[n]; } }
      // content addressing (write key) on old memory
      float kn = 0.f;
      for (int e = 0; e < CELL; ++e) kn += wk[e] * wk[e];
      kn = sqrtf(kn);
      float sc[N_CELLS], smx = -1e30f;
      for (int n = 0; n < N_CELLS; ++n) {
        float d = 0.f, mn2 = 0.f;
        for (int e = 0; e < CELL; ++e) { float mv = S.mem[m][n][e]; d += wk[e] * mv; mn2 += mv * mv; }
        sc[n] = d / (kn * sqrtf(mn2) + 1e-6f) * wstr;
        smx = fmaxf(smx, sc[n]);
      }
      float ssum = 0.f;
      for (int n = 0; n < N_CELLS; ++n) { sc[n] = __expf(sc[n] - smx); ssum += sc[n]; }
      for (int n = 0; n < N_CELLS; ++n) {
        float cw = sc[n] / ssum;
        S.ww[m][n] = gwt * (ga * alloc[n] + (1.f - ga) * cw);
      }
    }
    __syncthreads();

    // ======== parallel memory write + link update ========
    for (int idx = tid; idx < BTILE * N_CELLS * CELL; idx += 256) {
      int m = idx / (N_CELLS * CELL); int r2 = idx % (N_CELLS * CELL);
      int n = r2 / CELL, e = r2 % CELL;
      float wn = S.ww[m][n];
      S.mem[m][n][e] = S.mem[m][n][e] * (1.f - wn * S.er[m][e]) + wn * S.wvv[m][e];
    }
    for (int idx = tid; idx < BTILE * N_CELLS * N_CELLS; idx += 256) {
      int m = idx / (N_CELLS * N_CELLS); int r2 = idx % (N_CELLS * N_CELLS);
      int i2 = r2 / N_CELLS, j2 = r2 % N_CELLS;
      float wi = S.ww[m][i2], wj = S.ww[m][j2];
      float l = (1.f - wi - wj) * S.link[m][i2][j2] + wi * S.prec[m][j2];
      S.link[m][i2][j2] = (i2 == j2) ? 0.f : l;
    }
    __syncthreads();

    // ======== stage B: precedence, temporal links, read ========
    if (tid < BTILE) {
      int m = tid;
      float wsum = 0.f;
      for (int n = 0; n < N_CELLS; ++n) wsum += S.ww[m][n];
      for (int n = 0; n < N_CELLS; ++n) S.prec[m][n] = (1.f - wsum) * S.prec[m][n] + S.ww[m][n];
      float fwd[N_CELLS], bwd[N_CELLS];
      for (int i2 = 0; i2 < N_CELLS; ++i2) {
        float f = 0.f, bb = 0.f;
        for (int n = 0; n < N_CELLS; ++n) {
          f  += S.rw[m][n] * S.link[m][i2][n];
          bb += S.rw[m][n] * S.link[m][n][i2];
        }
        fwd[i2] = f; bwd[i2] = bb;
      }
      // content addressing (read key) on new memory
      float kn = 0.f;
      for (int e = 0; e < CELL; ++e) kn += rk[e] * rk[e];
      kn = sqrtf(kn);
      float sc[N_CELLS], smx = -1e30f;
      for (int n = 0; n < N_CELLS; ++n) {
        float d = 0.f, mn2 = 0.f;
        for (int e = 0; e < CELL; ++e) { float mv = S.mem[m][n][e]; d += rk[e] * mv; mn2 += mv * mv; }
        sc[n] = d / (kn * sqrtf(mn2) + 1e-6f) * rs;
        smx = fmaxf(smx, sc[n]);
      }
      float ssum = 0.f;
      for (int n = 0; n < N_CELLS; ++n) { sc[n] = __expf(sc[n] - smx); ssum += sc[n]; }
      for (int n = 0; n < N_CELLS; ++n)
        S.rw[m][n] = md0 * bwd[n] + md1 * (sc[n] / ssum) + md2 * fwd[n];
      for (int e = 0; e < CELL; ++e) {
        float a = 0.f;
        for (int n = 0; n < N_CELLS; ++n) a += S.rw[m][n] * S.mem[m][n][e];
        S.rv[m][e] = a;
      }
    }
    __syncthreads();

    // ======== final-step readout: y = [ctrl|rv] @ W_out^T + b_out ========
    if (t == T_STEPS - 1 && tid < BTILE * OUT_DIM) {
      int m = tid / OUT_DIM, od = tid % OUT_DIM;
      float a = b_out[od];
      for (int k = 0; k < HID; ++k)  a += (float)S.ch[m][k] * W_out[od * (HID + CELL) + k];
      for (int k = 0; k < CELL; ++k) a += S.rv[m][k] * W_out[od * (HID + CELL) + HID + k];
      out[(size_t)(bbase + m) * OUT_DIM + od] = a;
    }
    __syncthreads();
  }
}

// ---------------- host launch ------------------------------------------------------
extern "C" void kernel_launch(void* const* d_in, const int* in_sizes, int n_in,
                              void* d_out, int out_size, void* d_ws, size_t ws_size,
                              hipStream_t stream) {
  const float* x     = (const float*)d_in[0];
  const float* W_ih0 = (const float*)d_in[1];
  const float* W_hh0 = (const float*)d_in[2];
  const float* b0    = (const float*)d_in[3];
  const float* W_ih1 = (const float*)d_in[4];
  const float* W_hh1 = (const float*)d_in[5];
  const float* b1    = (const float*)d_in[6];
  const float* W_if  = (const float*)d_in[7];
  const float* b_if  = (const float*)d_in[8];
  const float* W_out = (const float*)d_in[9];
  const float* b_out = (const float*)d_in[10];
  _Float16* wp = (_Float16*)d_ws;
  (void)n_in; (void)out_size; (void)ws_size;

  auto pack = [&](const float* W, int off, int N, int K, int NT, int KT) {
    int total = NT * KT * 512;
    prepack_kernel<<<(total + 255) / 256, 256, 0, stream>>>(W, wp + off, N, K, NT, KT);
  };
  pack(W_ih0, OFF_IH0, 512, 30, 32, 1);
  pack(W_hh0, OFF_HH0, 512, 128, 32, 4);
  pack(W_ih1, OFF_IH1, 512, 128, 32, 4);
  pack(W_hh1, OFF_HH1, 512, 128, 32, 4);
  pack(W_if,  OFF_IF,  IFACE, 128, 6, 4);

  int B = in_sizes[0] / (T_STEPS * IN_DIM);
  dnc_persistent<<<B / BTILE, 256, 0, stream>>>(x, b0, b1, b_if, W_out, b_out, wp,
                                                (float*)d_out);
}
